// SeparableBernoulliSigmoidSimGLM_26448408609569
// MI455X (gfx1250) — matrix operand order repaired and checked by reading, not compiled, not run on hardware
//
#include <hip/hip_runtime.h>

typedef __attribute__((ext_vector_type(16))) _Float16 v16h;
typedef __attribute__((ext_vector_type(8)))  float    v8f;

#define B_    128
#define P_    2048
#define NB_   2000
#define NF_   250
#define NC_   16
#define R_    128
#define LS_   1750   // NB - NF
#define KP_   256    // padded window length (K for WMMA, 8 chunks of 32)
#define TB_   16     // time-block size (steps per WMMA base matmul)
#define NBLK_ 110    // ceil(LS_/TB_)
#define TPB_  256
#define NW_   8      // waves per workgroup (wave32)

__global__ __launch_bounds__(TPB_) void glm_sim_kernel(
    const float* __restrict__ stim_spat,   // (B,P)
    const float* __restrict__ stim_time,   // (NB)
    const float* __restrict__ init_sec,    // (B,NF)
    const float* __restrict__ cells,       // (B,NC,NB)
    const float* __restrict__ spat_filt,   // (P)
    const float* __restrict__ spat_bias,   // (1)
    const float* __restrict__ stf,         // (NF)
    const float* __restrict__ fbf,         // (NF)
    const float* __restrict__ cpf,         // (NC,NF)
    const float* __restrict__ u,           // (LS,B,R)
    float* __restrict__ out)               // (B,R,NB)
{
    __shared__ float    s_gensig[LS_ + 16];              // gensig_piece[b, 0..1749]
    __shared__ float    s_f[KP_];                        // feedback filter, zero-padded
    __shared__ unsigned s_mask[R_][8];                   // 256-bit window bitmask per repeat row
    __shared__ __align__(16) float s_base[NW_][16][16];  // WMMA base result [row][step]
    __shared__ __align__(16) float s_u[2][NW_][16][16];  // double-buffered u [step][row]
    __shared__ float    s_cf[NC_ * NF_];                 // coupling filter
    __shared__ float    s_red[TPB_];
    __shared__ unsigned s_m0[8];
    __shared__ float    s_drive;

    const int b    = blockIdx.x;
    const int tid  = threadIdx.x;
    const int lane = tid & 31;
    const int wv   = tid >> 5;
    const int rb   = wv * 16;          // this wave's repeat-row base
    const int n    = lane & 15;        // WMMA M (A) / N (B,D) index for this lane
    const int kh   = (lane >> 4) & 1;  // lane-half (selects K sub-range per ISA layout)

    // ---------------- setup: spatial drive (block reduction) ----------------
    {
        float a = 0.f;
        for (int i = tid; i < P_; i += TPB_) a += stim_spat[b * P_ + i] * spat_filt[i];
        s_red[tid] = a;
        __syncthreads();
        for (int s = TPB_ / 2; s > 0; s >>= 1) {
            if (tid < s) s_red[tid] += s_red[tid + s];
            __syncthreads();
        }
        if (tid == 0) s_drive = s_red[0];
    }

    // ---------------- stage filters in LDS ----------------
    for (int i = tid; i < NC_ * NF_; i += TPB_) s_cf[i] = cpf[i];
    for (int i = tid; i < KP_; i += TPB_)       s_f[i]  = (i < NF_) ? fbf[i] : 0.f;

    // ---------------- initial window bitmask (same for all repeats) ----------------
    if (tid < 8) {
        unsigned w = 0;
        for (int i = 0; i < 32; ++i) {
            int idx = tid * 32 + i;
            if (idx < NF_ && init_sec[b * NF_ + idx] != 0.f) w |= (1u << i);
        }
        s_m0[tid] = w;
    }
    __syncthreads();

    const float bias  = spat_bias[0];
    const float drive = s_drive;

    // ---------------- gensig[b,t] = drive*filt_time[t] + bias + coup[b,t] ----------------
    for (int t = tid; t < LS_; t += TPB_) {
        float ft = 0.f;
        for (int k = 0; k < NF_; ++k) ft += stim_time[t + k] * stf[k];
        float cp = 0.f;
        for (int c = 0; c < NC_; ++c) {
            const float* cr = &cells[(b * NC_ + c) * NB_ + t];
            const float* fr = &s_cf[c * NF_];
            for (int k = 0; k < NF_; ++k) cp += cr[k] * fr[k];
        }
        s_gensig[t] = drive * ft + bias + cp;
    }

    // replicate initial mask to all 128 repeat rows
    if (tid < R_) {
        for (int w = 0; w < 8; ++w) s_mask[tid][w] = s_m0[w];
    }

    // ---------------- emit w0 section of output: out[b,r,0:NF] = init[b,:] ----------------
    for (int i = tid; i < R_ * NF_; i += TPB_) {
        int r = i / NF_, k = i - r * NF_;
        out[(b * R_ + r) * NB_ + k] = init_sec[b * NF_ + k];
    }

    // ---------------- constant B operand: F[i][j] = f[i-j] (i>=j), 8 K-chunks ----------------
    // B 32x16 f16, contiguous K order used here (A matches): lane holds column n; lanes 0-15 K=0..15, lanes 16-31 K=16..31;
    // VGPR v holds K=2v,2v+1  ->  vector elems 2v/2v+1.
    v16h bfrag[8];
#pragma unroll
    for (int q = 0; q < 8; ++q) {
#pragma unroll
        for (int v = 0; v < 8; ++v) {
            int K = q * 32 + kh * 16 + 2 * v;
#pragma unroll
            for (int h = 0; h < 2; ++h) {
                int i = K + h;
                float val = (i >= n && (i - n) < NF_) ? fbf[i - n] : 0.f;
                bfrag[q][2 * v + h] = (_Float16)val;
            }
        }
    }
    __syncthreads();

    // ---- async staging of one 16-step u block into LDS buffer `nb` ----
    // 2 x B128 instructions per wave: lane covers 4 consecutive repeats of one step.
#define ASYNC_STAGE(tb, nb)                                                      \
    {                                                                            \
        const int t0n = (tb) * TB_;                                              \
        _Pragma("unroll")                                                        \
        for (int i = 0; i < 2; ++i) {                                            \
            int flat = i * 128 + lane * 4;                                       \
            int j = flat >> 4, rw = flat & 15;                                   \
            int tt = t0n + j;                                                    \
            if (tt > LS_ - 1) tt = LS_ - 1; /* clamp: garbage lands unused */    \
            const float* gp = u + ((size_t)tt * (B_ * R_) + b * R_ + rb + rw);   \
            unsigned ldsa = (unsigned)(size_t)&s_u[nb][wv][j][rw];               \
            unsigned long long ga = (unsigned long long)(size_t)gp;              \
            asm volatile("global_load_async_to_lds_b128 %0, %1, off"             \
                         :: "v"(ldsa), "v"(ga) : "memory");                      \
        }                                                                        \
    }

    // prologue: stage block 0
    ASYNC_STAGE(0, 0)

    // ---------------- main blocked recurrence ----------------
    for (int blk = 0; blk < NBLK_; ++blk) {
        const int t0  = blk * TB_;
        const int jn  = (LS_ - t0 < TB_) ? (LS_ - t0) : TB_;
        const int buf = blk & 1;

        // kick off async staging of the NEXT block into the other buffer
        if (blk + 1 < NBLK_) ASYNC_STAGE(blk + 1, buf ^ 1)

        // base(16 rows x 16 steps) = Window(16x256, f16 0/1) x F(256x16, f16), f32 acc
        // A 16x32 f16 layout: lane holds row M=n; per chunk q all K bits live in mask word q;
        // intra-chunk offset = kh*16 + v*2.
        v8f acc = {};
#pragma unroll
        for (int q = 0; q < 8; ++q) {
            unsigned word = s_mask[rb + n][q];
            v16h af;
#pragma unroll
            for (int v = 0; v < 8; ++v) {
                int off = (kh << 4) + (v << 1);
                af[2 * v]     = ((word >> off) & 1u)       ? (_Float16)1.0f : (_Float16)0.0f;
                af[2 * v + 1] = ((word >> (off + 1)) & 1u) ? (_Float16)1.0f : (_Float16)0.0f;
            }
            acc = __builtin_amdgcn_wmma_f32_16x16x32_f16(
                false, af, false, bfrag[q], (short)0, acc, false, false);
        }

        // spill D: lane holds column n, rows kh*8 + d
#pragma unroll
        for (int d = 0; d < 8; ++d) s_base[wv][kh * 8 + d][n] = acc[d];

        // wait for THIS block's async u (in-order completion; <=2 leaves only
        // the batch just issued for blk+1 outstanding)
        if (blk + 1 < NBLK_) asm volatile("s_wait_asynccnt 0x2" ::: "memory");
        else                 asm volatile("s_wait_asynccnt 0x0" ::: "memory");

        // sequential 16 steps; lane < 16 owns one repeat row
        if (lane < 16) {
            const int row = lane;
            // batch-preload everything the serial chain needs into registers
            float gb[TB_], ureg[TB_];
#pragma unroll
            for (int j = 0; j < TB_; ++j) ureg[j] = s_u[buf][wv][j][row];
#pragma unroll
            for (int j = 0; j < TB_; ++j) gb[j] = s_gensig[t0 + j] + s_base[wv][row][j];

            unsigned sb = 0;  // spikes emitted inside this block
            for (int j = 0; j < jn; ++j) {
                // intra-block correction: sum_{m<j, spike} f[NF - j + m]
                float corr = 0.f;
                unsigned bits = sb;
                while (bits) {
                    int m = __ffs(bits) - 1;
                    bits &= bits - 1;
                    corr += s_f[NF_ - j + m];
                }
                float g = gb[j] + corr;
                float p = __builtin_amdgcn_rcpf(1.0f + __expf(-g));
                if (ureg[j] < p) sb |= (1u << j);
            }

            // write spikes (offset is always even -> aligned float2 stores)
            const int ob = (b * R_ + rb + row) * NB_ + NF_ + t0;
            if (jn == TB_) {
#pragma unroll
                for (int q = 0; q < 8; ++q) {
                    float2 o;
                    o.x = (float)((sb >> (2 * q)) & 1u);
                    o.y = (float)((sb >> (2 * q + 1)) & 1u);
                    *(float2*)(out + ob + 2 * q) = o;
                }
                // shift window mask down 16 bits, insert new spikes at bit 234 (word 7, bit 10)
                unsigned* mw = s_mask[rb + row];
                unsigned w[8];
#pragma unroll
                for (int k = 0; k < 8; ++k) w[k] = mw[k];
#pragma unroll
                for (int k = 0; k < 7; ++k) mw[k] = (w[k] >> 16) | (w[k + 1] << 16);
                mw[7] = (w[7] >> 16) | (sb << 10);
            } else {
                for (int j = 0; j < jn; ++j) out[ob + j] = (float)((sb >> j) & 1u);
            }
        }
    }
#undef ASYNC_STAGE
}

extern "C" void kernel_launch(void* const* d_in, const int* in_sizes, int n_in,
                              void* d_out, int out_size, void* d_ws, size_t ws_size,
                              hipStream_t stream) {
    (void)in_sizes; (void)n_in; (void)out_size; (void)d_ws; (void)ws_size;
    glm_sim_kernel<<<B_, TPB_, 0, stream>>>(
        (const float*)d_in[0],   // batched_stim_spat
        (const float*)d_in[1],   // stim_time
        (const float*)d_in[2],   // batched_initial_spike_section
        (const float*)d_in[3],   // batched_coupling_cells
        (const float*)d_in[4],   // spatial_filter
        (const float*)d_in[5],   // spatial_bias
        (const float*)d_in[6],   // stim_timecourse_filter
        (const float*)d_in[7],   // feedback_filter
        (const float*)d_in[8],   // coupling_filter
        (const float*)d_in[9],   // u
        (float*)d_out);
}
